// ScannedRNN_7756710936902
// MI455X (gfx1250) — matrix operand order repaired and checked by reading, not compiled, not run on hardware
//
#include <hip/hip_runtime.h>

typedef __attribute__((ext_vector_type(16))) __bf16 v16bf;
typedef __attribute__((ext_vector_type(8)))  __bf16 v8bf;
typedef __attribute__((ext_vector_type(8)))  float  v8f;

#define T_STEPS 512
#define BATCH   128
#define HID     512
#define HID3    1536

__device__ __forceinline__ unsigned short f2bf_rne(float f) {
  unsigned int u = __float_as_uint(f);
  unsigned int r = u + 0x7FFFu + ((u >> 16) & 1u);
  return (unsigned short)(r >> 16);
}

__device__ __forceinline__ float sigmoid_fast(float x) {
  // 1 / (1 + e^-x) via v_exp_f32 + v_rcp_f32
  return __builtin_amdgcn_rcpf(1.0f + __expf(-x));
}

__device__ __forceinline__ float tanh_fast(float x) {
  // tanh(x) = 1 - 2 / (e^{2x} + 1)
  return 1.0f - 2.0f * __builtin_amdgcn_rcpf(__expf(2.0f * x) + 1.0f);
}

union Frag16 { v16bf v; v8bf h[2]; };

// ---------------- conversion kernels ----------------

__global__ void cvt_x_kernel(const float* __restrict__ x,
                             unsigned short* __restrict__ xb, int n) {
  int i = blockIdx.x * blockDim.x + threadIdx.x;
  if (i < n) xb[i] = f2bf_rne(x[i]);
}

// W [K=HID][N=HID3] row-major  ->  WT [N=HID3][K=HID] bf16 row-major
__global__ void cvt_wT_kernel(const float* __restrict__ W,
                              unsigned short* __restrict__ WT) {
  int i = blockIdx.x * blockDim.x + threadIdx.x;   // over HID3*HID
  int n = i >> 9;            // / HID
  int k = i & (HID - 1);
  WT[i] = f2bf_rne(W[(size_t)k * HID3 + n]);
}

// ---------------- big GEMM: gi = Xb @ Wi + bi ----------------
// M = T*B = 65536, N = 1536, K = 512.  Block = 4 waves, tile 64(M) x 64(N).
// Massive grid occupancy hides memory latency here; HBM-bound on gi writes.

__global__ void __launch_bounds__(128)
gemm_gi_kernel(const unsigned short* __restrict__ Xb,   // [M][HID]  bf16
               const unsigned short* __restrict__ WiT,  // [HID3][HID] bf16
               const float* __restrict__ bi,            // [HID3]
               float* __restrict__ gi)                  // [M][HID3] f32
{
  const int lane = threadIdx.x & 31;
  const int wv   = threadIdx.x >> 5;
  const int l16  = lane & 15;
  const int hs   = lane >> 4;

  const int m0 = blockIdx.x * 64 + wv * 16;
  const int n0 = blockIdx.y * 64;

  v8f acc[4];
  #pragma unroll
  for (int i = 0; i < 4; ++i) acc[i] = (v8f){0,0,0,0,0,0,0,0};

  const unsigned short* arow = Xb + (size_t)(m0 + l16) * HID + 16 * hs;
  const unsigned short* brow = WiT + (size_t)(n0 + l16) * HID + 16 * hs;

  #pragma unroll
  for (int k0 = 0; k0 < HID; k0 += 32) {
    Frag16 a;
    a.h[0] = *(const v8bf*)(arow + k0);
    a.h[1] = *(const v8bf*)(arow + k0 + 8);
    #pragma unroll
    for (int nt = 0; nt < 4; ++nt) {
      Frag16 b;
      b.v = *(const v16bf*)(brow + (size_t)(nt * 16) * HID + k0);
      acc[nt] = __builtin_amdgcn_wmma_f32_16x16x32_bf16(
          false, a.v, false, b.v, (short)0, acc[nt], false, false);
    }
  }

  #pragma unroll
  for (int nt = 0; nt < 4; ++nt) {
    int n = n0 + nt * 16 + l16;
    float bias = bi[n];
    #pragma unroll
    for (int v = 0; v < 8; ++v) {
      int m = m0 + v + 8 * hs;          // C layout: M = vgpr + 8*half
      gi[(size_t)m * HID3 + n] = acc[nt][v] + bias;
    }
  }
}

// ---------------- sequential GRU scan ----------------
// Grid = B/16 = 8 workgroups (independent batch slices), 8 waves each.
// Wave w owns h-columns [64w, 64w+64): 12 WMMA tiles (4 r, 4 z, 4 n slices).
// h in LDS (fp32 + bf16 copies, plus an all-zero row 16 used to apply the
// reset mask at consumption time).  WhT streams from L2 each step with a
// 4-deep software-pipelined B-fragment ring; step body fully unrolled so
// all addressing is base-pointer + immediate offset.

__global__ void __launch_bounds__(256)
gru_scan_kernel(const float* __restrict__ gi,            // [T*B][HID3]
                const int* __restrict__ resets,          // [T*B]
                const float* __restrict__ h0,            // [B][HID]
                const unsigned short* __restrict__ WhT,  // [HID3][HID] bf16
                const float* __restrict__ bhn,           // [HID]
                float* __restrict__ ys)                  // [T*B][HID]
{
  const int lane = threadIdx.x & 31;
  const int wv   = threadIdx.x >> 5;     // 0..7
  const int l16  = lane & 15;
  const int hs   = lane >> 4;
  const int b0   = blockIdx.x * 16;

  __shared__ __align__(16) float          hf[17][HID];        // row 16 = zeros
  __shared__ __align__(16) unsigned short hb[17][HID + 8];    // row 16 = zeros

  for (int idx = threadIdx.x; idx < 17 * HID; idx += 256) {
    int m = idx >> 9, j = idx & (HID - 1);
    float v = (m < 16) ? h0[(size_t)(b0 + m) * HID + j] : 0.0f;
    hf[m][j] = v;
    hb[m][j] = f2bf_rne(v);
  }
  __syncthreads();

  const int colbase = wv * 64;
  // Per-lane B base: row (colbase + l16) of WhT, K-chunk for this lane half.
  const unsigned short* Bbase = WhT + (size_t)(colbase + l16) * HID + 16 * hs;

  for (int t = 0; t < T_STEPS; ++t) {
    // ---- reset mask (applied at consumption; no LDS rewrite) ----
    const int rflagA = resets[t * BATCH + b0 + l16];
    const unsigned short* Arow = &hb[rflagA ? 16 : l16][16 * hs];

    v8f acc[12];
    #pragma unroll
    for (int i = 0; i < 12; ++i) acc[i] = (v8f){0,0,0,0,0,0,0,0};

    // ---- gh = h @ Wh : 192 WMMAs, 4-deep B prefetch ring ----
    Frag16 bq[4];
    #pragma unroll
    for (int p = 0; p < 4; ++p)            // it = 0..3 -> k0=0, tiles 0..3 (g=0)
      bq[p].v = *(const v16bf*)(Bbase + (size_t)(p * 16) * HID);

    Frag16 a, an;
    a.h[0] = *(const v8bf*)(Arow);
    a.h[1] = *(const v8bf*)(Arow + 8);

    #pragma unroll
    for (int it = 0; it < 192; ++it) {
      const int k0  = (it / 12) * 32;
      const int idx = it % 12;
      if (idx == 0 && k0 + 32 < HID) {     // one-ahead A fragment (from LDS)
        an.h[0] = *(const v8bf*)(Arow + k0 + 32);
        an.h[1] = *(const v8bf*)(Arow + k0 + 40);
      }
      acc[idx] = __builtin_amdgcn_wmma_f32_16x16x32_bf16(
          false, a.v, false, bq[it & 3].v, (short)0, acc[idx], false, false);
      if (it + 4 < 192) {                  // refill ring slot for it+4
        const int it2  = it + 4;
        const int k2   = (it2 / 12) * 32;
        const int idx2 = it2 % 12;
        bq[it & 3].v = *(const v16bf*)(
            Bbase + (size_t)(((idx2 >> 2) * HID + (idx2 & 3) * 16)) * HID + k2);
      }
      if (idx == 11) a = an;
    }

    // ---- per-row reset flags for the z*h term ----
    int rref[8];
    #pragma unroll
    for (int v = 0; v < 8; ++v)
      rref[v] = resets[t * BATCH + b0 + v + 8 * hs];

    // ---- gate math (registers only) ----
    float hnew[4][8];
    #pragma unroll
    for (int i = 0; i < 4; ++i) {
      int j = colbase + i * 16 + l16;
      float bh = bhn[j];
      #pragma unroll
      for (int v = 0; v < 8; ++v) {
        int m = v + 8 * hs;
        const float* gir = gi + (size_t)(t * BATCH + b0 + m) * HID3;
        float r  = sigmoid_fast(gir[j]       + acc[i][v]);
        float z  = sigmoid_fast(gir[HID + j] + acc[4 + i][v]);
        float nn = tanh_fast(gir[2 * HID + j] + r * (acc[8 + i][v] + bh));
        float ho = rref[v] ? 0.0f : hf[m][j];
        hnew[i][v] = (1.0f - z) * nn + z * ho;
      }
    }
    __syncthreads();   // all waves done reading hb/hf for this step

    #pragma unroll
    for (int i = 0; i < 4; ++i) {
      int j = colbase + i * 16 + l16;
      #pragma unroll
      for (int v = 0; v < 8; ++v) {
        int m = v + 8 * hs;
        float hv = hnew[i][v];
        hf[m][j] = hv;
        hb[m][j] = f2bf_rne(hv);
        ys[(size_t)(t * BATCH + b0 + m) * HID + j] = hv;
      }
    }
    __syncthreads();   // writes visible before next step
  }
}

// ---------------- host launcher ----------------

extern "C" void kernel_launch(void* const* d_in, const int* in_sizes, int n_in,
                              void* d_out, int out_size, void* d_ws, size_t ws_size,
                              hipStream_t stream) {
  const float* ins    = (const float*)d_in[0];   // [T,B,H]
  const int*   resets = (const int*)d_in[1];     // [T,B]
  const float* h0     = (const float*)d_in[2];   // [B,H]
  const float* Wi     = (const float*)d_in[3];   // [H,3H]
  const float* Wh     = (const float*)d_in[4];   // [H,3H]
  const float* bi     = (const float*)d_in[5];   // [3H]
  const float* bhn    = (const float*)d_in[6];   // [H]
  float* ys = (float*)d_out;

  char* ws = (char*)d_ws;
  size_t off = 0;
  auto take = [&](size_t bytes) -> void* {
    void* p = ws + off;
    off += (bytes + 255) & ~(size_t)255;
    return p;
  };
  unsigned short* Xb  = (unsigned short*)take((size_t)T_STEPS * BATCH * HID * 2);
  unsigned short* WiT = (unsigned short*)take((size_t)HID3 * HID * 2);
  unsigned short* WhT = (unsigned short*)take((size_t)HID3 * HID * 2);
  float*          gi  = (float*)take((size_t)T_STEPS * BATCH * HID3 * 4);
  (void)ws_size; (void)in_sizes; (void)n_in; (void)out_size;

  int nx = T_STEPS * BATCH * HID;
  cvt_x_kernel<<<(nx + 255) / 256, 256, 0, stream>>>(ins, Xb, nx);
  cvt_wT_kernel<<<(HID3 * HID) / 256, 256, 0, stream>>>(Wi, WiT);
  cvt_wT_kernel<<<(HID3 * HID) / 256, 256, 0, stream>>>(Wh, WhT);

  dim3 g1((T_STEPS * BATCH) / 64, HID3 / 64);
  gemm_gi_kernel<<<g1, 128, 0, stream>>>(Xb, WiT, bi, gi);

  gru_scan_kernel<<<BATCH / 16, 256, 0, stream>>>(gi, resets, h0, WhT, bhn, ys);
}
